// Reading_memory_5282809774415
// MI455X (gfx1250) — matrix operand order repaired and checked by reading, not compiled, not run on hardware
//
#include <hip/hip_runtime.h>

typedef __attribute__((ext_vector_type(16))) _Float16 v16h;
typedef __attribute__((ext_vector_type(8)))  _Float16 v8h;
typedef __attribute__((ext_vector_type(8)))  float    v8f;
typedef __attribute__((ext_vector_type(4)))  int      v4i;

#define HID   100
#define DL    5
#define NT    7        // N tiles (112 padded cols)
#define KC1   4        // K chunks for W_mem  (K=100+bias -> 128)
#define KC2   7        // K chunks for W_sub  (K=200+bias -> 224)
#define KC3   10       // K chunks for W_gate (K=300+bias -> 320)
#define ITEMS 16
#define ROWS  (ITEMS*DL)   // 80
#define MT    (ROWS/16)    // 5 M tiles

// packed weight image (dword units): per (ntile,kchunk) block = 256 dwords
#define WMEM_DW  0
#define WSUB_DW  (NT*KC1*256)              // 7168
#define WGATE_DW (WSUB_DW + NT*KC2*256)    // 19712
#define WTOT_DW  (WGATE_DW + NT*KC3*256)   // 37632

// LDS byte layout (all activation staging in f16 -> A frags are 2x ds_load_b128, no cvt)
#define OFF_W      0
#define OFF_MEMH   (WTOT_DW*4)                        // 150528 : f16 [80][128]
#define MEMH_S     128
#define OFF_ACTH   (OFF_MEMH + ROWS*MEMH_S*2)         // 171008 : f16 [80][224]
#define ACTH_S     224
#define OFF_GATEH  (OFF_ACTH + ROWS*ACTH_S*2)         // 206848 : f16 [80][320]
#define GATEH_S    320
#define OFF_M4A    (OFF_GATEH + ROWS*GATEH_S*2)       // 258048 : f16 [80][112]
#define ATT_S      112
#define OFF_S4A    (OFF_M4A + ROWS*ATT_S*2)           // 275968 : f16 [80][112]
#define LDS_BYTES  (OFF_S4A + ROWS*ATT_S*2)           // 293888 <= 320KB WGP LDS

#if __has_builtin(__builtin_amdgcn_global_load_async_to_lds_b128)
#define USE_ASYNC_COPY 1
#if __has_builtin(__builtin_amdgcn_s_wait_asynccnt)
#define WAIT_ASYNC() __builtin_amdgcn_s_wait_asynccnt(0)
#else
#define WAIT_ASYNC() asm volatile("s_wait_asynccnt 0" ::: "memory")
#endif
#else
#define USE_ASYNC_COPY 0
#endif

// ---------------- weight repack: f32 [K][N] -> f16 B-fragment image + bias row ----
__global__ void pack_weights_kernel(const float* Wm, const float* bm,
                                    const float* Ws, const float* bs,
                                    const float* Wg, const float* bg,
                                    unsigned int* wsout) {
    int idx = blockIdx.x * blockDim.x + threadIdx.x;
    if (idx >= WTOT_DW) return;
    const float* W; const float* bias; int kreal, kcn, local;
    if (idx < WSUB_DW)       { W = Wm; bias = bm; kreal = 100; kcn = KC1; local = idx; }
    else if (idx < WGATE_DW) { W = Ws; bias = bs; kreal = 200; kcn = KC2; local = idx - WSUB_DW; }
    else                     { W = Wg; bias = bg; kreal = 300; kcn = KC3; local = idx - WGATE_DW; }
    int block  = local >> 8;
    int within = local & 255;
    int lane = within >> 3;
    int j    = within & 7;
    int ntile = block / kcn;
    int kch   = block % kcn;
    // B 32x16 f16 frag in this code's contiguous k order (as load_a16): lane<16 -> K 0..15, lane>=16 -> K 16..31; dword j = K {2j,2j+1}
    int k0 = kch * 32 + (lane >> 4) * 16 + 2 * j;
    int n  = ntile * 16 + (lane & 15);
    float v0 = 0.f, v1 = 0.f;
    if (n < HID) {
        if (k0 < kreal)       v0 = W[k0 * HID + n];
        else if (k0 == kreal) v0 = bias[n];
        int k1 = k0 + 1;
        if (k1 < kreal)       v1 = W[k1 * HID + n];
        else if (k1 == kreal) v1 = bias[n];
    }
    _Float16 h0 = (_Float16)v0, h1 = (_Float16)v1;
    unsigned int u = (unsigned int)__builtin_bit_cast(unsigned short, h0)
                   | ((unsigned int)__builtin_bit_cast(unsigned short, h1) << 16);
    wsout[idx] = u;
}

// ---------------- helpers ---------------------------------------------------------
__device__ __forceinline__ unsigned int pk2(float a, float b) {
    _Float16 h0 = (_Float16)a, h1 = (_Float16)b;
    return (unsigned int)__builtin_bit_cast(unsigned short, h0)
         | ((unsigned int)__builtin_bit_cast(unsigned short, h1) << 16);
}

// branch-free transcendentals: v_exp_f32 / v_rcp_f32, no EXEC divergence
__device__ __forceinline__ float fast_tanh(float x) {
    float e = __expf(2.f * x);                         // inf for large x, 0 for very negative
    return 1.f - 2.f * __builtin_amdgcn_rcpf(e + 1.f); // -> +1 / -1 saturation naturally
}
__device__ __forceinline__ float fast_sigmoid(float x) {
    return __builtin_amdgcn_rcpf(1.f + __expf(-x));
}

__device__ __forceinline__ v16h load_a16(const _Float16* rowp, int kc, int lane) {
    // A 16x32 f16 frag, this code's contiguous k order (matches B pack): lane<16 -> K 0..15, lane>=16 -> K 16..31
    int kb = kc * 32 + (lane >> 4) * 16;
    v8h lo = *(const v8h*)(rowp + kb);        // ds_load_b128
    v8h hi = *(const v8h*)(rowp + kb + 8);    // ds_load_b128
    return __builtin_shufflevector(lo, hi, 0,1,2,3,4,5,6,7,8,9,10,11,12,13,14,15);
}

__device__ __forceinline__ v16h load_b_frag(const unsigned int* wbase, int blk, int lane) {
    return *(const v16h*)(wbase + blk * 256 + lane * 8);   // 2x ds_load_b128, 32B aligned
}

// ---------------- fused kernel ----------------------------------------------------
__launch_bounds__(256, 1)
__global__ void fused_reading_memory_kernel(const float* __restrict__ sub_emb,
                                            const float* __restrict__ memory,
                                            const int*   __restrict__ sub_len,
                                            const float* __restrict__ sub_raw,
                                            const unsigned int* __restrict__ wpacked,
                                            float* __restrict__ out, int B) {
    extern __shared__ char smem[];
    unsigned int* sW      = (unsigned int*)(smem + OFF_W);
    _Float16*     sMemH   = (_Float16*)(smem + OFF_MEMH);
    _Float16*     sActH   = (_Float16*)(smem + OFF_ACTH);
    _Float16*     sGateH  = (_Float16*)(smem + OFF_GATEH);
    _Float16*     sM4A    = (_Float16*)(smem + OFF_M4A);
    _Float16*     sS4A    = (_Float16*)(smem + OFF_S4A);
    unsigned int* sMemH32 = (unsigned int*)(smem + OFF_MEMH);
    unsigned int* sActH32 = (unsigned int*)(smem + OFF_ACTH);
    unsigned int* sGateH32= (unsigned int*)(smem + OFF_GATEH);

    const int tid  = threadIdx.x;
    const int wv   = tid >> 5;
    const int lane = tid & 31;
    const long itemBase = (long)blockIdx.x * ITEMS;

    // ---- phase 0a: weight image -> LDS (async DMA path, bypasses VGPRs) ----
#if USE_ASYNC_COPY
    for (int i = tid; i < WTOT_DW / 4; i += 256) {
        __builtin_amdgcn_global_load_async_to_lds_b128(
            (v4i*)(wpacked + i * 4), (v4i*)(sW + i * 4), 0, 0);
    }
#else
    {
        const float4* src = (const float4*)wpacked;
        float4* dst = (float4*)sW;
        for (int i = tid; i < WTOT_DW / 4; i += 256) dst[i] = src[i];
    }
#endif

    // ---- phase 0b: activation staging, f32 global -> packed f16 LDS (converted once) ----
    // memory rows: f16 [80][128], col 100 = 1.0 (bias row), 101..127 = 0
    for (int i = tid; i < ROWS * (MEMH_S / 2); i += 256) {
        int r = i / (MEMH_S / 2), c2 = i % (MEMH_S / 2), c = c2 * 2;
        long gi = itemBase + r / DL; int l = r % DL;
        float v0 = 0.f, v1 = 0.f;
        if (gi < B) {
            if (c < HID) { const float* p = memory + (gi * DL + l) * HID + c; v0 = p[0]; v1 = p[1]; }
            else if (c == HID) v0 = 1.0f;
        }
        sMemH32[i] = pk2(v0, v1);
    }
    // [sub_emb | raw] rows: f16 [80][224], col 200 = 1.0, 201..223 = 0
    // sub_emb halves also duplicated into gate A buffer cols 0..99
    for (int i = tid; i < ROWS * (ACTH_S / 2); i += 256) {
        int r = i / (ACTH_S / 2), c2 = i % (ACTH_S / 2), c = c2 * 2;
        long gi = itemBase + r / DL; int l = r % DL;
        float v0 = 0.f, v1 = 0.f;
        if (gi < B) {
            long base = (gi * DL + l) * HID;
            if (c < HID)        { v0 = sub_emb[base + c];       v1 = sub_emb[base + c + 1]; }
            else if (c < 200)   { v0 = sub_raw[base + c - 100]; v1 = sub_raw[base + c - 99]; }
            else if (c == 200)  v0 = 1.0f;
        }
        unsigned int u = pk2(v0, v1);
        sActH32[i] = u;
        if (c < HID) sGateH32[r * (GATEH_S / 2) + c2] = u;   // gate_in[:,0:100] = sub_emb
    }
    // gate A buffer bias/pad cols 300..319
    for (int i = tid; i < ROWS * 10; i += 256) {
        int r = i / 10, c2 = i % 10;
        sGateH32[r * (GATEH_S / 2) + 150 + c2] = (c2 == 0) ? pk2(1.0f, 0.f) : 0u;
    }
#if USE_ASYNC_COPY
    WAIT_ASYNC();
#endif
    __syncthreads();

    // ---- phase 1: mem_4att = tanh(memory @ W_mem + b) ----
    for (int tile = wv; tile < MT * NT; tile += 8) {
        int mt = tile / NT, nt = tile % NT;
        const _Float16* rowp = sMemH + (mt * 16 + (lane & 15)) * MEMH_S;
        v8f acc = {};
#pragma unroll
        for (int kc = 0; kc < KC1; ++kc) {
            v16h a = load_a16(rowp, kc, lane);
            v16h b = load_b_frag(sW + WMEM_DW, nt * KC1 + kc, lane);
            acc = __builtin_amdgcn_wmma_f32_16x16x32_f16(false, a, false, b, (short)0, acc, false, false);
        }
        int col = nt * 16 + (lane & 15);
        int rb  = mt * 16 + ((lane >> 4) ? 8 : 0);
#pragma unroll
        for (int v = 0; v < 8; ++v)
            sM4A[(rb + v) * ATT_S + col] = (_Float16)fast_tanh(acc[v]);
    }

    // ---- phase 2: sub_4att = tanh([sub_emb|raw] @ W_sub + b) ----
    for (int tile = wv; tile < MT * NT; tile += 8) {
        int mt = tile / NT, nt = tile % NT;
        const _Float16* rowp = sActH + (mt * 16 + (lane & 15)) * ACTH_S;
        v8f acc = {};
#pragma unroll
        for (int kc = 0; kc < KC2; ++kc) {
            v16h a = load_a16(rowp, kc, lane);
            v16h b = load_b_frag(sW + WSUB_DW, nt * KC2 + kc, lane);
            acc = __builtin_amdgcn_wmma_f32_16x16x32_f16(false, a, false, b, (short)0, acc, false, false);
        }
        int col = nt * 16 + (lane & 15);
        int rb  = mt * 16 + ((lane >> 4) ? 8 : 0);
#pragma unroll
        for (int v = 0; v < 8; ++v)
            sS4A[(rb + v) * ATT_S + col] = (_Float16)fast_tanh(acc[v]);
    }
    __syncthreads();

    // ---- phase 3: 5x5 attention + softmax + sub_mem readout (VALU; ~1% of FLOPs) ----
    if (tid < ROWS) {
        int r = tid, item = r / DL, q = r % DL;
        long gi = itemBase + item;
        if (gi < B) {
            float att[DL];
            int len = sub_len[gi];
            if (q < len) {
                float mx = -1e30f;
#pragma unroll
                for (int k = 0; k < DL; ++k) {
                    float s = 0.f;
                    for (int h = 0; h < HID; ++h)
                        s += (float)sS4A[r * ATT_S + h] * (float)sM4A[(item * DL + k) * ATT_S + h];
                    att[k] = s; mx = fmaxf(mx, s);
                }
                float sum = 0.f;
#pragma unroll
                for (int k = 0; k < DL; ++k) { att[k] = __expf(att[k] - mx); sum += att[k]; }
                float inv = __builtin_amdgcn_rcpf(sum);
#pragma unroll
                for (int k = 0; k < DL; ++k) att[k] *= inv;
            } else {
#pragma unroll
                for (int k = 0; k < DL; ++k) att[k] = 0.f;   // zero_mask row
            }
            for (int h = 0; h < HID; ++h) {
                float sm = 0.f;
#pragma unroll
                for (int k = 0; k < DL; ++k)
                    sm += att[k] * (float)sMemH[(item * DL + k) * MEMH_S + h];
                float se = (float)sActH[r * ACTH_S + h];
                sGateH[r * GATEH_S + 100 + h] = (_Float16)sm;          // gate_in[:,100:200]
                sGateH[r * GATEH_S + 200 + h] = (_Float16)(sm * se);   // gate_in[:,200:300]
            }
        }
    }
    __syncthreads();

    // ---- phase 4: g = sigmoid(gate_in @ W_gate + b); out = (1-g)*sub_emb + g*sub_mem ----
    for (int tile = wv; tile < MT * NT; tile += 8) {
        int mt = tile / NT, nt = tile % NT;
        const _Float16* rowp = sGateH + (mt * 16 + (lane & 15)) * GATEH_S;
        v8f acc = {};
#pragma unroll
        for (int kc = 0; kc < KC3; ++kc) {
            v16h a = load_a16(rowp, kc, lane);
            v16h b = load_b_frag(sW + WGATE_DW, nt * KC3 + kc, lane);
            acc = __builtin_amdgcn_wmma_f32_16x16x32_f16(false, a, false, b, (short)0, acc, false, false);
        }
        int col = nt * 16 + (lane & 15);
        if (col < HID) {
            int rb = mt * 16 + ((lane >> 4) ? 8 : 0);
#pragma unroll
            for (int v = 0; v < 8; ++v) {
                int r = rb + v;
                long gi = itemBase + r / DL;
                if (gi < B) {
                    float g  = fast_sigmoid(acc[v]);
                    float se = (float)sActH[r * ACTH_S + col];
                    float sm = (float)sGateH[r * GATEH_S + 100 + col];
                    out[(gi * DL + (r % DL)) * HID + col] = (1.f - g) * se + g * sm;
                }
            }
        }
    }
}

extern "C" void kernel_launch(void* const* d_in, const int* in_sizes, int n_in,
                              void* d_out, int out_size, void* d_ws, size_t ws_size,
                              hipStream_t stream) {
    const float* sub_emb = (const float*)d_in[0];
    const float* memory  = (const float*)d_in[1];
    const int*   sub_len = (const int*)d_in[2];
    const float* sub_raw = (const float*)d_in[3];
    const float* Wm = (const float*)d_in[4];
    const float* bm = (const float*)d_in[5];
    const float* Ws = (const float*)d_in[6];
    const float* bs = (const float*)d_in[7];
    const float* Wg = (const float*)d_in[8];
    const float* bg = (const float*)d_in[9];
    int B = in_sizes[2];

    unsigned int* wpacked = (unsigned int*)d_ws;   // 150,528 bytes used
    pack_weights_kernel<<<(WTOT_DW + 255) / 256, 256, 0, stream>>>(Wm, bm, Ws, bs, Wg, bg, wpacked);

    int grid = (B + ITEMS - 1) / ITEMS;
    fused_reading_memory_kernel<<<grid, 256, LDS_BYTES, stream>>>(
        sub_emb, memory, sub_len, sub_raw, wpacked, (float*)d_out, B);
}
